// LstmModel_80951543595851
// MI455X (gfx1250) — matrix operand order repaired and checked by reading, not compiled, not run on hardware
//
#include <hip/hip_runtime.h>

typedef __attribute__((ext_vector_type(16))) _Float16 v16h;
typedef __attribute__((ext_vector_type(8)))  _Float16 v8h;
typedef __attribute__((ext_vector_type(8)))  float    v8f;
typedef __attribute__((ext_vector_type(4)))  float    v4f;
typedef int v4i __attribute__((vector_size(4 * sizeof(int))));   // int4, gcc-style

#define B_     2048
#define T_     256
#define F_     64
#define EMB_   16
#define UNITS_ 32
#define CAT_   48
#define TPW_   4      // M-tiles per wave in the embedding kernel

#if defined(__has_builtin)
#if __has_builtin(__builtin_amdgcn_global_load_async_to_lds_b128) && \
    __has_builtin(__builtin_amdgcn_s_wait_asynccnt)
#define ASYNC_E 1
#endif
#if __has_builtin(__builtin_amdgcn_tanhf)
#define HAVE_TANH 1
#endif
#endif

#if ASYNC_E
typedef __attribute__((address_space(1))) v4i* gptr_v4i;
typedef __attribute__((address_space(3))) v4i* lptr_v4i;
#endif

// Fast activations: avoid the IEEE div sequence (v_div_scale/.../v_div_fixup).
#if HAVE_TANH
__device__ __forceinline__ float tanh_fast(float x) {
    return __builtin_amdgcn_tanhf(x);                 // v_tanh_f32 (TRANS32)
}
__device__ __forceinline__ float sigf(float x) {
    return fmaf(0.5f, __builtin_amdgcn_tanhf(0.5f * x), 0.5f);
}
#else
__device__ __forceinline__ float sigf(float x) {
    return __builtin_amdgcn_rcpf(1.0f + __expf(-x));  // v_exp + v_rcp
}
__device__ __forceinline__ float tanh_fast(float x) {
    x = fminf(15.0f, fmaxf(-15.0f, x));
    float e = __expf(2.0f * x);
    return (e - 1.0f) * __builtin_amdgcn_rcpf(e + 1.0f);
}
#endif

// ---------------------------------------------------------------------------
// Embedding: e[t, b, :] = sigmoid(x[b, t, :] @ W_emb + b_emb), stored fp16 in
// workspace as [T, B, EMB].  B fragments built once per block in LDS; each
// wave owns TPW_ consecutive 16-row M tiles of the flattened [B*T, 64] input.
// K=64 -> two v_wmma_f32_16x16x32_f16 per tile.
// ---------------------------------------------------------------------------
__global__ void emb_kernel(const float* __restrict__ x,
                           const float* __restrict__ W_emb,
                           const float* __restrict__ b_emb,
                           _Float16* __restrict__ e)
{
    __shared__ _Float16 lds_bf[2 * 32 * 16];   // [chunk][lane][16 halves] = 2 KB

    int lane = threadIdx.x & 31;
    int g    = lane >> 4;        // lane-half
    int lm   = lane & 15;

    // ---- one-time per block: per-lane B fragments of W_emb (64x16) --------
    // B layout (32x16 KxN): n = lm, K = chunk*32 + g*16 + i, contiguous in i.
    if (threadIdx.x < 32) {
        #pragma unroll
        for (int c = 0; c < 2; ++c) {
            _Float16* dst = &lds_bf[(c * 32 + lane) * 16];
            int kb = c * 32 + g * 16;
            #pragma unroll
            for (int i = 0; i < 16; ++i)
                dst[i] = (_Float16)W_emb[(kb + i) * EMB_ + lm];
        }
    }
    __syncthreads();

    v16h bm0 = *(const v16h*)&lds_bf[(0 * 32 + lane) * 16];
    v16h bm1 = *(const v16h*)&lds_bf[(1 * 32 + lane) * 16];
    float bias = b_emb[lm];

    int wave = (blockIdx.x * blockDim.x + threadIdx.x) >> 5;

    #pragma unroll
    for (int it = 0; it < TPW_; ++it) {
        long row0 = ((long)wave * TPW_ + it) * 16;
        const float* xr = x + (row0 + lm) * F_;

        // prefetch next tile's row for this lane (global_prefetch_b8)
        if (it + 1 < TPW_)
            __builtin_prefetch(xr + 16 * F_, 0, 0);

        // A fragments: contiguous k order, same as the B fragments:
        //   halves 0..7  -> K = kc + g*16 + i
        //   halves 8..15 -> K = kc + g*16 + 8 + (i-8)
        v16h a[2];
        #pragma unroll
        for (int c = 0; c < 2; ++c) {
            int kb0 = c * 32 + g * 16;
            int kb1 = c * 32 + g * 16 + 8;
            v4f p0 = *(const v4f*)(xr + kb0);
            v4f p1 = *(const v4f*)(xr + kb0 + 4);
            v4f p2 = *(const v4f*)(xr + kb1);
            v4f p3 = *(const v4f*)(xr + kb1 + 4);
            #pragma unroll
            for (int i = 0; i < 4; ++i) {
                a[c][i]      = (_Float16)p0[i];
                a[c][4 + i]  = (_Float16)p1[i];
                a[c][8 + i]  = (_Float16)p2[i];
                a[c][12 + i] = (_Float16)p3[i];
            }
        }

        v8f acc;
        #pragma unroll
        for (int i = 0; i < 8; ++i) acc[i] = bias;

        acc = __builtin_amdgcn_wmma_f32_16x16x32_f16(false, a[0], false, bm0,
                                                     (short)0, acc, false, false);
        acc = __builtin_amdgcn_wmma_f32_16x16x32_f16(false, a[1], false, bm1,
                                                     (short)0, acc, false, false);

        // D layout: vgpr r, lane -> (m = r + 8*g, n = lm). Flat row = b*T + t.
        #pragma unroll
        for (int r = 0; r < 8; ++r) {
            long row = row0 + r + 8 * g;
            long b   = row / T_;
            long t   = row - b * T_;
            e[(t * B_ + b) * EMB_ + lm] = (_Float16)sigf(acc[r]);
        }
    }
}

// ---------------------------------------------------------------------------
// LSTM scan: one wave owns 16 batch rows for all 256 steps.
// Per step: z = [h | e_t | 0pad] (16x64 f16), g = z @ W_gates (64x128 f16),
// 8 N-tiles x 2 k-chunks = 16 WMMAs.  W_gates pre-packed in LDS in
// fragment-native order.  e tile double-buffered in LDS via async-to-LDS
// copies (ASYNCcnt) so the global fetch is off the recurrence critical path.
// ---------------------------------------------------------------------------
__global__ void lstm_kernel(const _Float16* __restrict__ e,
                            const float* __restrict__ Wf, const float* __restrict__ Wi,
                            const float* __restrict__ Wc, const float* __restrict__ Wo,
                            const float* __restrict__ bf, const float* __restrict__ bi,
                            const float* __restrict__ bc, const float* __restrict__ bo,
                            const float* __restrict__ Wout, const float* __restrict__ bout,
                            float* __restrict__ out)
{
    __shared__ _Float16 lds_wg[16 * 32 * 16];   // [frag][lane][16 halves] = 16 KB
    __shared__ _Float16 lds_h[16 * 32];         // h tile [16][32] fp16 = 1 KB
#if ASYNC_E
    __shared__ _Float16 lds_e[2 * 16 * EMB_];   // double-buffered e tile = 1 KB
#endif

    int lane = threadIdx.x & 31;
    int g    = lane >> 4;
    int lm   = lane & 15;
    int b0   = blockIdx.x * 16;

    // ---- one-time: pack W_gates fragments into LDS (zero-pad K 48..63) ----
    #pragma unroll
    for (int f = 0; f < 16; ++f) {
        int tile  = f >> 1;
        int chunk = f & 1;
        int n     = tile * 16 + lm;          // global gate column 0..127
        int gate  = n >> 5;
        int col   = n & 31;
        const float* W = (gate == 0) ? Wf : (gate == 1) ? Wi : (gate == 2) ? Wc : Wo;
        _Float16* dst = &lds_wg[(f * 32 + lane) * 16];
        #pragma unroll
        for (int i = 0; i < 16; ++i) {
            int k = chunk * 32 + (i < 8 ? g * 8 + i : 16 + g * 8 + (i - 8));
            dst[i] = (k < CAT_) ? (_Float16)W[k * UNITS_ + col] : (_Float16)0.0f;
        }
    }
    for (int i = lane; i < 16 * 32; i += 32) lds_h[i] = (_Float16)0.0f;
    __syncthreads();

    // per-lane bias, folded into accumulator init
    float bias[8];
    #pragma unroll
    for (int tile = 0; tile < 8; ++tile) {
        int n    = tile * 16 + lm;
        int gate = n >> 5;
        int col  = n & 31;
        const float* bb = (gate == 0) ? bf : (gate == 1) ? bi : (gate == 2) ? bc : bo;
        bias[tile] = bb[col];
    }

    // per-lane e addresses: lane copies/reads its own 16 bytes of the tile
    int e_lds_off = lm * EMB_ + g * 8;                       // halves
#if ASYNC_E
    // issue async copy of the t=0 tile into buffer 0
    {
        const _Float16* src = e + ((long)0 * B_ + b0 + lm) * EMB_ + g * 8;
        __builtin_amdgcn_global_load_async_to_lds_b128(
            (gptr_v4i)src, (lptr_v4i)&lds_e[e_lds_off], 0, 0);
    }
#endif

    v8f c0 = {}, c1 = {};
    v8f h0 = {}, h1 = {};

    for (int t = 0; t < T_; ++t) {
        // ---- A chunk1 (K=32..63) = [e_t | zeros] -------------------------
        v16h a1;
#if ASYNC_E
        if (t + 1 < T_) {
            const _Float16* src = e + ((long)(t + 1) * B_ + b0 + lm) * EMB_ + g * 8;
            __builtin_amdgcn_global_load_async_to_lds_b128(
                (gptr_v4i)src,
                (lptr_v4i)&lds_e[((t + 1) & 1) * 256 + e_lds_off], 0, 0);
            __builtin_amdgcn_s_wait_asynccnt(1);   // buffer t ready (in-order)
        } else {
            __builtin_amdgcn_s_wait_asynccnt(0);
        }
        {
            v8h q = *(const v8h*)&lds_e[(t & 1) * 256 + e_lds_off];
            #pragma unroll
            for (int i = 0; i < 8; ++i) { a1[i] = q[i]; a1[8 + i] = (_Float16)0.0f; }
        }
#else
        {
            const _Float16* er = e + ((long)t * B_ + b0 + lm) * EMB_ + g * 8;
            v8h q = *(const v8h*)er;
            #pragma unroll
            for (int i = 0; i < 8; ++i) { a1[i] = q[i]; a1[8 + i] = (_Float16)0.0f; }
        }
        int tn = (t + 1 < T_) ? (t + 1) : t;
        __builtin_prefetch(e + ((long)tn * B_ + b0 + lm) * EMB_, 0, 0);
#endif

        // ---- A chunk0 (K=0..31) = h, from LDS (D-layout -> A-layout) -----
        v16h a0;
        {
            const _Float16* hr = &lds_h[lm * 32];
            v8h q0 = *(const v8h*)(hr + g * 8);
            v8h q1 = *(const v8h*)(hr + 16 + g * 8);
            #pragma unroll
            for (int i = 0; i < 8; ++i) { a0[i] = q0[i]; a0[8 + i] = q1[i]; }
        }

        // ---- gate GEMM: 8 N-tiles x 2 k-chunks ---------------------------
        v8f acc[8];
        #pragma unroll
        for (int tile = 0; tile < 8; ++tile) {
            #pragma unroll
            for (int i = 0; i < 8; ++i) acc[tile][i] = bias[tile];
            v16h bfr0 = *(const v16h*)&lds_wg[((tile * 2 + 0) * 32 + lane) * 16];
            v16h bfr1 = *(const v16h*)&lds_wg[((tile * 2 + 1) * 32 + lane) * 16];
            acc[tile] = __builtin_amdgcn_wmma_f32_16x16x32_f16(
                false, a0, false, bfr0, (short)0, acc[tile], false, false);
            acc[tile] = __builtin_amdgcn_wmma_f32_16x16x32_f16(
                false, a1, false, bfr1, (short)0, acc[tile], false, false);
        }

        // ---- gates: tiles 0,1=f  2,3=i  4,5=ch  6,7=o --------------------
        #pragma unroll
        for (int i = 0; i < 8; ++i) {
            float f0  = sigf(acc[0][i]);
            float i0  = sigf(acc[2][i]);
            float ch0 = tanh_fast(acc[4][i]);
            float o0  = sigf(acc[6][i]);
            float cn0 = f0 * c0[i] + i0 * ch0;
            c0[i] = cn0;
            h0[i] = o0 * tanh_fast(cn0);

            float f1  = sigf(acc[1][i]);
            float i1  = sigf(acc[3][i]);
            float ch1 = tanh_fast(acc[5][i]);
            float o1  = sigf(acc[7][i]);
            float cn1 = f1 * c1[i] + i1 * ch1;
            c1[i] = cn1;
            h1[i] = o1 * tanh_fast(cn1);
        }

        // ---- write h back to LDS (fp16) for next step's A fragment -------
        #pragma unroll
        for (int r = 0; r < 8; ++r) {
            int m = r + 8 * g;
            lds_h[m * 32 + lm]      = (_Float16)h0[r];
            lds_h[m * 32 + 16 + lm] = (_Float16)h1[r];
        }
        __syncthreads();   // single-wave WG: ordering fence only
    }

    // ---- output head: out[b] = sigmoid(h_last[b,:] @ W_out + b_out) ------
    if (lane < 16) {
        float acc = bout[0];
        #pragma unroll
        for (int k = 0; k < 32; ++k)
            acc += (float)lds_h[lane * 32 + k] * Wout[k];
        out[b0 + lane] = sigf(acc);
    }
}

// ---------------------------------------------------------------------------
extern "C" void kernel_launch(void* const* d_in, const int* in_sizes, int n_in,
                              void* d_out, int out_size, void* d_ws, size_t ws_size,
                              hipStream_t stream)
{
    const float* x     = (const float*)d_in[0];
    const float* W_emb = (const float*)d_in[1];
    const float* b_emb = (const float*)d_in[2];
    const float* W_f   = (const float*)d_in[3];
    const float* b_f   = (const float*)d_in[4];
    const float* W_i   = (const float*)d_in[5];
    const float* b_i   = (const float*)d_in[6];
    const float* W_c   = (const float*)d_in[7];
    const float* b_c   = (const float*)d_in[8];
    const float* W_o   = (const float*)d_in[9];
    const float* b_o   = (const float*)d_in[10];
    const float* W_out = (const float*)d_in[11];
    const float* b_out = (const float*)d_in[12];
    float* out = (float*)d_out;

    _Float16* e = (_Float16*)d_ws;   // [T, B, EMB] fp16 = 16 MB

    // 32768 M-tiles, 8 waves/block, 4 tiles/wave -> 1024 blocks
    emb_kernel<<<(B_ * T_ / 16) / (8 * TPW_), 256, 0, stream>>>(x, W_emb, b_emb, e);

    // one 32-thread wave per 16 batch rows -> 128 blocks
    lstm_kernel<<<B_ / 16, 32, 0, stream>>>(e,
                                            W_f, W_i, W_c, W_o,
                                            b_f, b_i, b_c, b_o,
                                            W_out, b_out, out);
}
